// Model_68410239090894
// MI455X (gfx1250) — matrix operand order repaired and checked by reading, not compiled, not run on hardware
//
#include <hip/hip_runtime.h>

typedef __attribute__((ext_vector_type(16))) _Float16 v16h;
typedef __attribute__((ext_vector_type(8)))  _Float16 v8h;
typedef __attribute__((ext_vector_type(8)))  float    v8f;

#define SLOPE 0.01f

// LDS layout (bytes), dynamic shared memory:
//   [0, 128K)            staged W0 (K=256 x N=256) f16, WMMA-B permuted
//   [128K, 160K)         staged W1 (K=256 x N=64)  f16, WMMA-B permuted
//   [160K + wave*8K, +8K) per-wave h stage (16 rows x 256 cols) f16, WMMA-A permuted
#define W0_OFF   0
#define W1_OFF   (128 * 1024)
#define H_OFF    (160 * 1024)
#define SMEM_SZ  (H_OFF + 8 * 8192)   // 229376 bytes <= 320KB WGP LDS

static __device__ __forceinline__ float lrelu(float x) {
    // x>=0: x >= 0.01x ; x<0: 0.01x > x  => leaky = max(x, 0.01x)
    return fmaxf(x, SLOPE * x);
}

// Stage weights (f32 global, row-major [K][N]) into LDS as f16 in the exact
// per-lane WMMA B layout: lane = (n%16) + 16*((k%16)/8); 16 contiguous f16
// per (tile j, kstep, lane) with t = (k%8) + 8*((k%32)/16).
static __device__ void stage_weights(const float* __restrict__ W0,
                                     const float* __restrict__ W1,
                                     char* smem, int tid) {
    _Float16* w0 = (_Float16*)(smem + W0_OFF);
    for (int idx = tid; idx < 256 * 256; idx += 256) {   // coalesced over n
        int k = idx >> 8, n = idx & 255;
        int lane = (n & 15) + (((k >> 3) & 1) << 4);
        int off = (((n >> 4) * 8 + (k >> 5)) * 32 + lane) * 16 + ((k & 7) | (((k >> 4) & 1) << 3));
        w0[off] = (_Float16)W0[idx];
    }
    _Float16* w1 = (_Float16*)(smem + W1_OFF);
    for (int idx = tid; idx < 256 * 64; idx += 256) {
        int k = idx >> 6, n = idx & 63;
        int lane = (n & 15) + (((k >> 3) & 1) << 4);
        int off = (((n >> 4) * 8 + (k >> 5)) * 32 + lane) * 16 + ((k & 7) | (((k >> 4) & 1) << 3));
        w1[off] = (_Float16)W1[idx];
    }
    __syncthreads();
}

// Read one 32x16 B tile (or 16x32 A tile from the h region): 16 contiguous
// f16 per lane -> two ds_load_b128.
static __device__ __forceinline__ v16h load_btile(const _Float16* base, int j, int ks, int lane) {
    const _Float16* p = base + ((j * 8 + ks) * 32 + lane) * 16;
    v8h lo = *(const v8h*)p;
    v8h hi = *(const v8h*)(p + 8);
    return __builtin_shufflevector(lo, hi, 0,1,2,3,4,5,6,7,8,9,10,11,12,13,14,15);
}

// Build one 16x32 f16 A tile from two concatenated f32 sources (each 128 wide,
// row-major). ISA A layout: lanes 0-15 hold M=lane, K = {k0..k0+7, k0+16..k0+23}
// with k0 = 0; lanes 16-31 same rows with k0 = 8.
static __device__ __forceinline__ v16h load_atile(const float* __restrict__ p0,
                                                  const float* __restrict__ p1,
                                                  int row, int kstep, int lane) {
    int k0 = kstep * 32 + ((lane & 16) ? 8 : 0);
    v16h a;
#pragma unroll
    for (int h = 0; h < 2; ++h) {
        int kb = k0 + 16 * h;   // multiple of 8 -> never straddles the 128 split
        const float* src = (kb < 128) ? (p0 + row * 128 + kb)
                                      : (p1 + row * 128 + (kb - 128));
        float4 f0 = *(const float4*)src;
        float4 f1 = *(const float4*)(src + 4);
        a[h*8+0] = (_Float16)f0.x;  a[h*8+1] = (_Float16)f0.y;
        a[h*8+2] = (_Float16)f0.z;  a[h*8+3] = (_Float16)f0.w;
        a[h*8+4] = (_Float16)f1.x;  a[h*8+5] = (_Float16)f1.y;
        a[h*8+6] = (_Float16)f1.z;  a[h*8+7] = (_Float16)f1.w;
    }
    return a;
}

// Scatter a C tile (16x16, cols [16*j0,16*j0+16) of the 256-wide h) into the
// per-wave LDS h region, permuted into the WMMA A-tile layout for layer 1.
static __device__ __forceinline__ void store_htile(_Float16* hbase, int j0, int lane, const v8f& v) {
    int q  = lane & 15;
    int t  = ((j0 & 1) << 3) | (q & 7);
    int ks = j0 >> 1;
    int rl_hi = (q >= 8) ? 16 : 0;
    int mb = (lane & 16) ? 8 : 0;      // C layout: lanes>=16 hold M=r+8
#pragma unroll
    for (int r = 0; r < 8; ++r) {
        int rl = (r + mb) + rl_hi;     // reader lane
        hbase[(ks * 32 + rl) * 16 + t] = (_Float16)v[r];
    }
}

// One wave computes a 16-row tile through the 256->256->64 LeakyReLU MLP.
static __device__ __forceinline__ void mlp_16rows(const v16h A[8],
                                                  const char* smem, _Float16* hreg,
                                                  const float* __restrict__ b0,
                                                  const float* __restrict__ b1,
                                                  int lane, v8f y[4]) {
    const _Float16* w0 = (const _Float16*)(smem + W0_OFF);
    const _Float16* w1 = (const _Float16*)(smem + W1_OFF);
    int q = lane & 15;

    // layer 0: [16x256] @ [256x256]
    for (int j = 0; j < 16; ++j) {
        v8f acc = {};
#pragma unroll
        for (int ks = 0; ks < 8; ++ks) {
            v16h b = load_btile(w0, j, ks, lane);
            acc = __builtin_amdgcn_wmma_f32_16x16x32_f16(false, A[ks], false, b,
                                                         (short)0, acc, false, false);
        }
        float bias = b0[j * 16 + q];
#pragma unroll
        for (int e = 0; e < 8; ++e) acc[e] = lrelu(acc[e] + bias);
        store_htile(hreg, j, lane, acc);
    }

    // layer 1: [16x256] @ [256x64]
    v8f acc1[4] = {};
    for (int ks = 0; ks < 8; ++ks) {
        v16h ap = load_btile(hreg, 0, ks, lane);   // A' from per-wave h region
#pragma unroll
        for (int j = 0; j < 4; ++j) {
            v16h b = load_btile(w1, j, ks, lane);
            acc1[j] = __builtin_amdgcn_wmma_f32_16x16x32_f16(false, ap, false, b,
                                                             (short)0, acc1[j], false, false);
        }
    }
#pragma unroll
    for (int j = 0; j < 4; ++j) {
        float bias = b1[j * 16 + q];
#pragma unroll
        for (int e = 0; e < 8; ++e) y[j][e] = lrelu(acc1[j][e] + bias);
    }
}

// ---- Kernel 1: shared ui MLP, writes ui_emb [B,64] f32 to workspace ----
__global__ __launch_bounds__(256) void ui_mlp_kernel(
    const float* __restrict__ u_emb, const float* __restrict__ i_emb,
    const float* __restrict__ W0, const float* __restrict__ b0,
    const float* __restrict__ W1, const float* __restrict__ b1,
    float* __restrict__ ui_out) {
    extern __shared__ char smem[];
    int tid = threadIdx.x, lane = tid & 31, wave = tid >> 5;
    int row_base = blockIdx.x * 128 + wave * 16;
    _Float16* hreg = (_Float16*)(smem + H_OFF + wave * 8192);

    stage_weights(W0, W1, smem, tid);

    v16h A[8];
    int row = row_base + (lane & 15);
#pragma unroll
    for (int ks = 0; ks < 8; ++ks) A[ks] = load_atile(u_emb, i_emb, row, ks, lane);

    v8f y[4];
    mlp_16rows(A, smem, hreg, b0, b1, lane, y);

    int q = lane & 15;
    int mb = (lane & 16) ? 8 : 0;
#pragma unroll
    for (int j = 0; j < 4; ++j)
#pragma unroll
        for (int r = 0; r < 8; ++r)
            ui_out[(row_base + r + mb) * 64 + j * 16 + q] = y[j][r];
}

// ---- Kernel 2: per-relation ao MLP (both relations), select by s, dot ----
__global__ __launch_bounds__(256) void ao_pred_kernel(
    const float* __restrict__ a_emb, const float* __restrict__ o_emb,
    const int* __restrict__ s,
    const float* __restrict__ W_ao0, const float* __restrict__ b_ao0,
    const float* __restrict__ W_ao1, const float* __restrict__ b_ao1,
    const float* __restrict__ ui_emb,
    float* __restrict__ pred) {
    extern __shared__ char smem[];
    int tid = threadIdx.x, lane = tid & 31, wave = tid >> 5;
    int row_base = blockIdx.x * 128 + wave * 16;   // flat row over (b, n), N=50
    _Float16* hreg = (_Float16*)(smem + H_OFF + wave * 8192);

    v16h A[8];
    int row = row_base + (lane & 15);
#pragma unroll
    for (int ks = 0; ks < 8; ++ks) A[ks] = load_atile(a_emb, o_emb, row, ks, lane);

    v8f y[2][4];
#pragma unroll
    for (int rel = 0; rel < 2; ++rel) {
        __syncthreads();                            // all waves done reading staged weights
        stage_weights(W_ao0 + rel * 256 * 256, W_ao1 + rel * 256 * 64, smem, tid);
        mlp_16rows(A, smem, hreg, b_ao0 + rel * 256, b_ao1 + rel * 64, lane, y[rel]);
    }

    // select relation per row, dot with ui_emb[b], reduce over 16-lane halves
    int q = lane & 15;
    int mb = (lane & 16) ? 8 : 0;
#pragma unroll
    for (int r = 0; r < 8; ++r) {
        int mrow = row_base + r + mb;
        int sv = s[mrow];
        int b = mrow / 50;
        float p = 0.f;
#pragma unroll
        for (int j = 0; j < 4; ++j) {
            float uiv = ui_emb[b * 64 + j * 16 + q];
            float yv  = (sv != 0) ? y[1][j][r] : y[0][j][r];
            p += yv * uiv;
        }
#pragma unroll
        for (int off = 1; off < 16; off <<= 1) p += __shfl_xor(p, off, 32);
        if (q == 0) pred[mrow] = p;                 // lane 0 -> m=r(+mb); lane 16 -> m=r+8
    }
}

extern "C" void kernel_launch(void* const* d_in, const int* in_sizes, int n_in,
                              void* d_out, int out_size, void* d_ws, size_t ws_size,
                              hipStream_t stream) {
    const float* u_emb = (const float*)d_in[0];
    const float* i_emb = (const float*)d_in[1];
    const float* a_emb = (const float*)d_in[2];
    const float* o_emb = (const float*)d_in[3];
    const int*   s     = (const int*)  d_in[4];
    const float* W_ao0 = (const float*)d_in[5];
    const float* b_ao0 = (const float*)d_in[6];
    const float* W_ao1 = (const float*)d_in[7];
    const float* b_ao1 = (const float*)d_in[8];
    const float* W_ui0 = (const float*)d_in[9];
    const float* b_ui0 = (const float*)d_in[10];
    const float* W_ui1 = (const float*)d_in[11];
    const float* b_ui1 = (const float*)d_in[12];

    float* ui_ws = (float*)d_ws;     // [4096*64] f32 intermediate
    float* pred  = (float*)d_out;    // [4096*50] f32

    (void)in_sizes; (void)n_in; (void)out_size; (void)ws_size;

    hipFuncSetAttribute((const void*)ui_mlp_kernel,
                        hipFuncAttributeMaxDynamicSharedMemorySize, SMEM_SZ);
    hipFuncSetAttribute((const void*)ao_pred_kernel,
                        hipFuncAttributeMaxDynamicSharedMemorySize, SMEM_SZ);

    // 4096 rows / 128 rows per WG = 32 blocks
    ui_mlp_kernel<<<32, 256, SMEM_SZ, stream>>>(u_emb, i_emb, W_ui0, b_ui0, W_ui1, b_ui1, ui_ws);
    // 4096*50 = 204800 rows / 128 = 1600 blocks
    ao_pred_kernel<<<1600, 256, SMEM_SZ, stream>>>(a_emb, o_emb, s, W_ao0, b_ao0,
                                                   W_ao1, b_ao1, ui_ws, pred);
}
